// TemT_89283780149429
// MI455X (gfx1250) — matrix operand order repaired and checked by reading, not compiled, not run on hardware
//
#include <hip/hip_runtime.h>
#include <hip/hip_bf16.h>
#include <math.h>

typedef __attribute__((ext_vector_type(16))) __bf16 v16bf;
typedef __attribute__((ext_vector_type(8)))  float  v8f;
typedef __attribute__((ext_vector_type(4)))  int    i32x4;
typedef unsigned short u16;
typedef unsigned int   u32;

#define BATCH 16
#define SEQ   512
#define DMODEL 768
#define BL (BATCH*SEQ)
#define NHEAD 8
#define DHEAD 96
#define DFFN  2048
#define DRNN  256

#if defined(__has_builtin)
# if __has_builtin(__builtin_amdgcn_global_load_async_to_lds_b128)
#  define USE_ASYNC_LDS 1
# endif
#endif

#ifdef USE_ASYNC_LDS
// signature (from compiler diagnostic): (i32x4 addrspace(1)*, i32x4 addrspace(3)*, Imm offset, Imm cpol)
#define ASYNC_CP16(gsrc, ldst)                                                         \
    __builtin_amdgcn_global_load_async_to_lds_b128(                                     \
        (__attribute__((address_space(1))) i32x4*)                                      \
            (__attribute__((address_space(1))) const void*)(const void*)(gsrc),         \
        (__attribute__((address_space(3))) i32x4*)                                      \
            (__attribute__((address_space(3))) void*)(void*)(ldst), 0, 0)
__device__ inline void wait_async0() {
# if __has_builtin(__builtin_amdgcn_s_wait_asynccnt)
    __builtin_amdgcn_s_wait_asynccnt(0);
# else
    asm volatile("s_wait_asynccnt 0x0" ::: "memory");
# endif
}
#endif

union FragB16 { v16bf v; u32 u[8]; };

__device__ inline u16 f2bf(float x) {
    u32 u = __float_as_uint(x);
    u += 0x7fffu + ((u >> 16) & 1u);          // round-to-nearest-even
    return (u16)(u >> 16);
}

__device__ inline v8f wmma_bf16(v16bf a, v16bf b, v8f c) {
    return __builtin_amdgcn_wmma_f32_16x16x32_bf16(
        false, a, false, b, (short)0, c, false, false);
}

// A fragment: 16x32 bf16 tile, row-major source (pairs along K contiguous).
__device__ inline v16bf load_frag_a(const u16* p, int ld, int lane) {
    int m  = lane & 15;
    int kb = (lane >> 4) * 8;
    const u16* row = p + m * ld;
    FragB16 f;
#pragma unroll
    for (int i = 0; i < 4; ++i) f.u[i]     = *(const u32*)(row + kb + 2*i);
#pragma unroll
    for (int i = 0; i < 4; ++i) f.u[4 + i] = *(const u32*)(row + 16 + kb + 2*i);
    return f.v;
}

// B fragment: 32x16 (KxN) from an N-major (N x K) tile.
__device__ inline v16bf load_frag_b(const u16* p, int ld, int lane) {
    int n  = lane & 15;
    int kb = (lane >> 4) * 8;
    const u16* row = p + n * ld;
    FragB16 f;
#pragma unroll
    for (int i = 0; i < 4; ++i) f.u[i]     = *(const u32*)(row + kb + 2*i);
#pragma unroll
    for (int i = 0; i < 4; ++i) f.u[4 + i] = *(const u32*)(row + 16 + kb + 2*i);
    return f.v;
}

// ------------------------------------------------------------ big GEMM (NT)
// C[M,N] = A[M,K]bf16 * B[N,K]bf16^T.  Requires M%128==0, N%128==0, K%32==0.
// 8 waves; block tile 128x128; wave tile 64x32 (4x2 WMMA); double-buffered LDS.
__global__ __launch_bounds__(256)
void k_gemm_nt_big(const u16* __restrict__ A, int lda,
                   const u16* __restrict__ B, int ldb,
                   float* Cf, u16* Cb, int ldc,
                   int K, const float* __restrict__ bias, int do_gelu)
{
    __shared__ u16 sA[2][128][32];
    __shared__ u16 sB[2][128][32];
    const int bm = blockIdx.x * 128, bn = blockIdx.y * 128;
    const int tid = threadIdx.x, wave = tid >> 5, lane = tid & 31;
    const int wr = (wave & 1) * 64;   // 2 row groups of 64
    const int wc = (wave >> 1) * 32;  // 4 col groups of 32

    const int r_st  = tid >> 2;          // 0..63  (two passes cover 128 rows)
    const int c8_st = (tid & 3) * 8;     // 0,8,16,24

    const u16* gA0 = A + (size_t)(bm + r_st)      * lda + c8_st;
    const u16* gA1 = A + (size_t)(bm + r_st + 64) * lda + c8_st;
    const u16* gB0 = B + (size_t)(bn + r_st)      * ldb + c8_st;
    const u16* gB1 = B + (size_t)(bn + r_st + 64) * ldb + c8_st;

    v8f acc[4][2];
#pragma unroll
    for (int i = 0; i < 4; ++i)
#pragma unroll
        for (int j = 0; j < 2; ++j)
#pragma unroll
            for (int e = 0; e < 8; ++e) acc[i][j][e] = 0.f;

    const int nk = K >> 5;

    // ---- stage K-step 0 into buffer 0 ----
#ifdef USE_ASYNC_LDS
    ASYNC_CP16(gA0, &sA[0][r_st][c8_st]);
    ASYNC_CP16(gA1, &sA[0][r_st + 64][c8_st]);
    ASYNC_CP16(gB0, &sB[0][r_st][c8_st]);
    ASYNC_CP16(gB1, &sB[0][r_st + 64][c8_st]);
    wait_async0();
#else
    {
        uint4 a0 = *(const uint4*)gA0;
        uint4 a1 = *(const uint4*)gA1;
        uint4 b0 = *(const uint4*)gB0;
        uint4 b1 = *(const uint4*)gB1;
        *(uint4*)&sA[0][r_st][c8_st]      = a0;
        *(uint4*)&sA[0][r_st + 64][c8_st] = a1;
        *(uint4*)&sB[0][r_st][c8_st]      = b0;
        *(uint4*)&sB[0][r_st + 64][c8_st] = b1;
    }
#endif
    __syncthreads();

    for (int kt = 0; kt < nk; ++kt) {
        const int cur = kt & 1;
        const bool pre = (kt + 1 < nk);
        const int k0n = (kt + 1) << 5;

#ifdef USE_ASYNC_LDS
        if (pre) {
            ASYNC_CP16(gA0 + k0n, &sA[cur ^ 1][r_st][c8_st]);
            ASYNC_CP16(gA1 + k0n, &sA[cur ^ 1][r_st + 64][c8_st]);
            ASYNC_CP16(gB0 + k0n, &sB[cur ^ 1][r_st][c8_st]);
            ASYNC_CP16(gB1 + k0n, &sB[cur ^ 1][r_st + 64][c8_st]);
        }
#else
        uint4 pa0, pa1, pb0, pb1;
        if (pre) {
            pa0 = *(const uint4*)(gA0 + k0n);
            pa1 = *(const uint4*)(gA1 + k0n);
            pb0 = *(const uint4*)(gB0 + k0n);
            pb1 = *(const uint4*)(gB1 + k0n);
        }
#endif

        v16bf af[4], bf[2];
#pragma unroll
        for (int i = 0; i < 4; ++i) af[i] = load_frag_a(&sA[cur][wr + 16 * i][0], 32, lane);
#pragma unroll
        for (int j = 0; j < 2; ++j) bf[j] = load_frag_b(&sB[cur][wc + 16 * j][0], 32, lane);
#pragma unroll
        for (int i = 0; i < 4; ++i)
#pragma unroll
            for (int j = 0; j < 2; ++j)
                acc[i][j] = wmma_bf16(af[i], bf[j], acc[i][j]);

#ifdef USE_ASYNC_LDS
        wait_async0();
#else
        if (pre) {
            *(uint4*)&sA[cur ^ 1][r_st][c8_st]      = pa0;
            *(uint4*)&sA[cur ^ 1][r_st + 64][c8_st] = pa1;
            *(uint4*)&sB[cur ^ 1][r_st][c8_st]      = pb0;
            *(uint4*)&sB[cur ^ 1][r_st + 64][c8_st] = pb1;
        }
#endif
        __syncthreads();
    }

#pragma unroll
    for (int i = 0; i < 4; ++i)
#pragma unroll
        for (int j = 0; j < 2; ++j)
#pragma unroll
            for (int e = 0; e < 8; ++e) {
                int row = bm + wr + 16 * i + (lane >> 4) * 8 + e;
                int col = bn + wc + 16 * j + (lane & 15);
                float v = acc[i][j][e];
                if (bias)    v += bias[col];
                if (do_gelu) v = 0.5f * v * (1.f + erff(v * 0.70710678f));
                if (Cf) Cf[(size_t)row * ldc + col] = v;
                if (Cb) Cb[(size_t)row * ldc + col] = f2bf(v);
            }
}

// ------------------------------------------------------- generic GEMM (NT)
// Guarded version for small M (head GEMMs).
#define TM 128
#define TN 64
__global__ __launch_bounds__(256)
void k_gemm_nt(const u16* __restrict__ A, int lda,
               const u16* __restrict__ B, int ldb,
               float* Cf, u16* Cb, int ldc,
               int M, int N, int K,
               const float* __restrict__ bias, int do_gelu)
{
    __shared__ u16 sA[TM][32];
    __shared__ u16 sB[TN][32];
    const int bm = blockIdx.x * TM, bn = blockIdx.y * TN;
    const int tid = threadIdx.x, wave = tid >> 5, lane = tid & 31;
    const int wr = (wave & 3) * 32;
    const int wc = (wave >> 2) * 32;

    v8f acc[2][2];
#pragma unroll
    for (int a = 0; a < 2; ++a)
#pragma unroll
        for (int b = 0; b < 2; ++b)
#pragma unroll
            for (int i = 0; i < 8; ++i) acc[a][b][i] = 0.f;

    const int r_st  = tid >> 2;
    const int c8_st = (tid & 3) * 8;
    const int grA0 = bm + r_st,  grA1 = bm + r_st + 64,  grB = bn + r_st;
    const int caA0 = grA0 < M - 1 ? grA0 : M - 1;
    const int caA1 = grA1 < M - 1 ? grA1 : M - 1;
    const int caB  = grB  < N - 1 ? grB  : N - 1;

    for (int k0 = 0; k0 < K; k0 += 32) {
        uint4 a0 = *(const uint4*)(A + (size_t)caA0 * lda + k0 + c8_st);
        uint4 a1 = *(const uint4*)(A + (size_t)caA1 * lda + k0 + c8_st);
        uint4 b0 = *(const uint4*)(B + (size_t)caB  * ldb + k0 + c8_st);
        if (grA0 >= M) { a0.x = a0.y = a0.z = a0.w = 0u; }
        if (grA1 >= M) { a1.x = a1.y = a1.z = a1.w = 0u; }
        if (grB  >= N) { b0.x = b0.y = b0.z = b0.w = 0u; }
        *(uint4*)&sA[r_st][c8_st]      = a0;
        *(uint4*)&sA[r_st + 64][c8_st] = a1;
        *(uint4*)&sB[r_st][c8_st]      = b0;
        __syncthreads();
        v16bf fa0 = load_frag_a(&sA[wr][0],      32, lane);
        v16bf fa1 = load_frag_a(&sA[wr + 16][0], 32, lane);
        v16bf fb0 = load_frag_b(&sB[wc][0],      32, lane);
        v16bf fb1 = load_frag_b(&sB[wc + 16][0], 32, lane);
        acc[0][0] = wmma_bf16(fa0, fb0, acc[0][0]);
        acc[0][1] = wmma_bf16(fa0, fb1, acc[0][1]);
        acc[1][0] = wmma_bf16(fa1, fb0, acc[1][0]);
        acc[1][1] = wmma_bf16(fa1, fb1, acc[1][1]);
        __syncthreads();
    }

#pragma unroll
    for (int tm = 0; tm < 2; ++tm)
#pragma unroll
        for (int tn = 0; tn < 2; ++tn)
#pragma unroll
            for (int i = 0; i < 8; ++i) {
                int row = bm + wr + tm * 16 + (lane >> 4) * 8 + i;
                int col = bn + wc + tn * 16 + (lane & 15);
                if (row < M && col < N) {
                    float v = acc[tm][tn][i];
                    if (bias)    v += bias[col];
                    if (do_gelu) v = 0.5f * v * (1.f + erff(v * 0.70710678f));
                    if (Cf) Cf[(size_t)row * ldc + col] = v;
                    if (Cb) Cb[(size_t)row * ldc + col] = f2bf(v);
                }
            }
}

// ---------------------------------------------------------------- attention
// One block per (b, h, 16-query rows). 128 threads = 4 waves.
__global__ __launch_bounds__(128)
void k_attn(const u16* __restrict__ Q, const u16* __restrict__ Kd,
            const u16* __restrict__ Vd, const float* __restrict__ t,
            u16* __restrict__ O)
{
    __shared__ float sS[16][SEQ];
    __shared__ u16   sP[16][SEQ];
    __shared__ u16   sVt[DHEAD][34];       // padded even ld for dword reads
    const int b = blockIdx.z, h = blockIdx.y, q0 = blockIdx.x * 16;
    const int tid = threadIdx.x, wave = tid >> 5, lane = tid & 31;
    const int LD = DMODEL;
    const u16* Qp = Q + ((size_t)(b * SEQ + q0)) * LD + h * DHEAD;
    const u16* Kp = Kd + ((size_t)b * SEQ) * LD + h * DHEAD;
    const u16* Vp = Vd + ((size_t)b * SEQ) * LD + h * DHEAD;

    // ---- S = Q K^T  (K dim = 96 -> 3 WMMA steps) ----
    v16bf aq[3];
#pragma unroll
    for (int kt = 0; kt < 3; ++kt) aq[kt] = load_frag_a(Qp + kt * 32, LD, lane);

    for (int j = 0; j < 8; ++j) {                 // 8 key tiles per wave
        int nt = wave * 8 + j;
        v8f acc;
#pragma unroll
        for (int i = 0; i < 8; ++i) acc[i] = 0.f;
#pragma unroll
        for (int kt = 0; kt < 3; ++kt) {
            v16bf bk = load_frag_b(Kp + (size_t)(nt * 16) * LD + kt * 32, LD, lane);
            acc = wmma_bf16(aq[kt], bk, acc);
        }
#pragma unroll
        for (int i = 0; i < 8; ++i) {
            int r = (lane >> 4) * 8 + i;
            int c = nt * 16 + (lane & 15);
            sS[r][c] = acc[i];
        }
    }
    __syncthreads();

    // ---- mask + softmax: wave w owns rows w, w+4, w+8, w+12 ----
    const float scale = 0.102062072f;             // 1/sqrt(96)
    for (int rr = wave; rr < 16; rr += 4) {
        int qrow = q0 + rr;
        float vals[16];
        float mx = -3.0e38f;
#pragma unroll
        for (int cc = 0; cc < 16; ++cc) {
            int c = cc * 32 + lane;
            float s = sS[rr][c] * scale;
            bool msk = (c > qrow) || (t[b * SEQ + c] == 0.f);
            s = msk ? -1.0e9f : s;
            vals[cc] = s;
            mx = fmaxf(mx, s);
        }
        for (int o = 16; o; o >>= 1) mx = fmaxf(mx, __shfl_xor(mx, o, 32));
        float sum = 0.f;
#pragma unroll
        for (int cc = 0; cc < 16; ++cc) { vals[cc] = expf(vals[cc] - mx); sum += vals[cc]; }
        for (int o = 16; o; o >>= 1) sum += __shfl_xor(sum, o, 32);
        float inv = 1.f / sum;
#pragma unroll
        for (int cc = 0; cc < 16; ++cc) sP[rr][cc * 32 + lane] = f2bf(vals[cc] * inv);
    }
    __syncthreads();

    // ---- O = P V ----
    v8f oacc0, oacc1;
#pragma unroll
    for (int i = 0; i < 8; ++i) { oacc0[i] = 0.f; oacc1[i] = 0.f; }

    for (int k0 = 0; k0 < SEQ; k0 += 32) {
        // stage V tile transposed: b64 loads, unpack to sVt[d][key]
#pragma unroll
        for (int s = 0; s < 6; ++s) {
            int idx = tid + s * 128;              // 0..767
            int kk = idx / 24;                    // 0..31
            int d4 = (idx % 24) * 4;              // 0..92
            uint2 w = *(const uint2*)(Vp + (size_t)(k0 + kk) * LD + d4);
            sVt[d4 + 0][kk] = (u16)(w.x & 0xffffu);
            sVt[d4 + 1][kk] = (u16)(w.x >> 16);
            sVt[d4 + 2][kk] = (u16)(w.y & 0xffffu);
            sVt[d4 + 3][kk] = (u16)(w.y >> 16);
        }
        __syncthreads();
        v16bf ap = load_frag_a(&sP[0][k0], SEQ, lane);
        v16bf bv0 = load_frag_b(&sVt[wave * 16][0], 34, lane);
        oacc0 = wmma_bf16(ap, bv0, oacc0);
        if (wave < 2) {
            v16bf bv1 = load_frag_b(&sVt[(4 + wave) * 16][0], 34, lane);
            oacc1 = wmma_bf16(ap, bv1, oacc1);
        }
        __syncthreads();
    }
    {
#pragma unroll
        for (int i = 0; i < 8; ++i) {
            int r = (lane >> 4) * 8 + i;
            int d = wave * 16 + (lane & 15);
            O[((size_t)(b * SEQ + q0 + r)) * LD + h * DHEAD + d] = f2bf(oacc0[i]);
        }
        if (wave < 2) {
#pragma unroll
            for (int i = 0; i < 8; ++i) {
                int r = (lane >> 4) * 8 + i;
                int d = (4 + wave) * 16 + (lane & 15);
                O[((size_t)(b * SEQ + q0 + r)) * LD + h * DHEAD + d] = f2bf(oacc1[i]);
            }
        }
    }
}

// ---------------------------------------------------------------- elementwise
__global__ void k_cumsum_t(const float* __restrict__ et, const float* __restrict__ npm,
                           float* __restrict__ t)
{
    int b = threadIdx.x;
    if (b >= BATCH) return;
    float s = 0.f;
    for (int l = 0; l < SEQ; ++l) {
        s += et[b * SEQ + l];
        t[b * SEQ + l] = s * npm[b * SEQ + l];
    }
}

__global__ void k_prep_x(const float* __restrict__ xin, const float* __restrict__ t,
                         const float* __restrict__ posv, const float* __restrict__ npm,
                         float* __restrict__ xout, u16* __restrict__ xb)
{
    int idx = blockIdx.x * 256 + threadIdx.x;
    if (idx >= BL * DMODEL) return;
    int d = idx % DMODEL, row = idx / DMODEL;
    float arg = t[row] / posv[d];
    float tem = (((d & 1) == 0) ? sinf(arg) : cosf(arg)) * npm[row];
    float v = xin[idx] + tem;
    xout[idx] = v;
    xb[idx] = f2bf(v);
}

__global__ __launch_bounds__(256)
void k_resid_ln(const float* __restrict__ x, const float* __restrict__ g,
                const float* __restrict__ bias, const float* __restrict__ gamma,
                const float* __restrict__ beta, const float* __restrict__ npm,
                float* __restrict__ xout, u16* __restrict__ xb)
{
    __shared__ float red[256];
    int row = blockIdx.x, tid = threadIdx.x;
    float v[3], s = 0.f;
#pragma unroll
    for (int j = 0; j < 3; ++j) {
        int d = j * 256 + tid;
        v[j] = x[(size_t)row * DMODEL + d] + g[(size_t)row * DMODEL + d] + bias[d];
        s += v[j];
    }
    red[tid] = s; __syncthreads();
    for (int o = 128; o; o >>= 1) { if (tid < o) red[tid] += red[tid + o]; __syncthreads(); }
    float mean = red[0] * (1.f / DMODEL);
    __syncthreads();
    float s2 = 0.f;
#pragma unroll
    for (int j = 0; j < 3; ++j) { float dvi = v[j] - mean; s2 += dvi * dvi; }
    red[tid] = s2; __syncthreads();
    for (int o = 128; o; o >>= 1) { if (tid < o) red[tid] += red[tid + o]; __syncthreads(); }
    float rstd = rsqrtf(red[0] * (1.f / DMODEL) + 1e-6f);
    float nm = npm[row];
#pragma unroll
    for (int j = 0; j < 3; ++j) {
        int d = j * 256 + tid;
        float y = (gamma[d] * (v[j] - mean) * rstd + beta[d]) * nm;
        xout[(size_t)row * DMODEL + d] = y;
        xb[(size_t)row * DMODEL + d] = f2bf(y);
    }
}

__global__ void k_conv_w(const float* __restrict__ in, u16* __restrict__ out,
                         int K, int N, int transpose)
{
    int idx = blockIdx.x * 256 + threadIdx.x;
    if (idx >= K * N) return;
    if (transpose) {
        int n = idx / K, k = idx % K;
        out[idx] = f2bf(in[(size_t)k * N + n]);   // out is N x K
    } else {
        out[idx] = f2bf(in[idx]);
    }
}

__global__ void k_bias_sum(const float* a, const float* b, float* o, int n)
{
    int i = blockIdx.x * 256 + threadIdx.x;
    if (i < n) o[i] = a[i] + b[i];
}

__global__ void k_x0_extract(const float* __restrict__ x, u16* __restrict__ x0b)
{
    int idx = blockIdx.x * 256 + threadIdx.x;
    if (idx >= BATCH * DMODEL) return;
    int b = idx / DMODEL, d = idx % DMODEL;
    x0b[idx] = f2bf(x[((size_t)b * SEQ) * DMODEL + d]);
}

__global__ void k_lstm(const float* __restrict__ gates, u16* __restrict__ hhb)
{
    int idx = blockIdx.x * 256 + threadIdx.x;
    if (idx >= BATCH * DRNN) return;
    int b = idx / DRNN, j = idx % DRNN;
    float ig = gates[b * 4 * DRNN + j];
    float gg = gates[b * 4 * DRNN + 2 * DRNN + j];
    float og = gates[b * 4 * DRNN + 3 * DRNN + j];
    float c  = (1.f / (1.f + expf(-ig))) * tanhf(gg);
    float hh = (1.f / (1.f + expf(-og))) * tanhf(c);
    hhb[idx] = f2bf(hh);
}

__global__ void k_unsort(const float* __restrict__ enc, const int* __restrict__ sidx,
                         float* __restrict__ out)
{
    int idx = blockIdx.x * 256 + threadIdx.x;
    if (idx >= BATCH * 256) return;
    int i = idx / 256, c = idx % 256;
    int j = 0;
    for (int k = 0; k < BATCH; ++k) if (sidx[k] == i) j = k;
    out[idx] = enc[j * 256 + c];
}

// ---------------------------------------------------------------- host
extern "C" void kernel_launch(void* const* d_in, const int* in_sizes, int n_in,
                              void* d_out, int out_size, void* d_ws, size_t ws_size,
                              hipStream_t stream)
{
    (void)in_sizes; (void)n_in; (void)out_size; (void)ws_size;
    const float* event_type = (const float*)d_in[0];
    const float* event_time = (const float*)d_in[1];
    const float* npm        = (const float*)d_in[2];
    const int*   sort_idx   = (const int*)d_in[3];
    const float* Wq   = (const float*)d_in[4];
    const float* Wk   = (const float*)d_in[5];
    const float* Wv   = (const float*)d_in[6];
    const float* Wo   = (const float*)d_in[7];
    const float* bo   = (const float*)d_in[8];
    const float* ln1g = (const float*)d_in[9];
    const float* ln1b = (const float*)d_in[10];
    const float* W1   = (const float*)d_in[11];
    const float* b1   = (const float*)d_in[12];
    const float* W2   = (const float*)d_in[13];
    const float* b2   = (const float*)d_in[14];
    const float* ln2g = (const float*)d_in[15];
    const float* ln2b = (const float*)d_in[16];
    const float* Wih  = (const float*)d_in[17];
    const float* bih  = (const float*)d_in[18];
    const float* bhh  = (const float*)d_in[20];
    const float* Wproj = (const float*)d_in[21];
    const float* bproj = (const float*)d_in[22];
    const float* Wg   = (const float*)d_in[23];
    const float* bg   = (const float*)d_in[24];
    const float* posv = (const float*)d_in[25];

    char* ws = (char*)d_ws;
    size_t off = 0;
    auto alloc = [&](size_t bytes) -> size_t {
        size_t o = off; off += (bytes + 255) & ~(size_t)255; return o;
    };
    const size_t szXf = (size_t)BL * DMODEL * 4;
    const size_t szXb = (size_t)BL * DMODEL * 2;
    size_t o_t    = alloc((size_t)BL * 4);
    size_t o_x    = alloc(szXf);
    size_t o_xb   = alloc(szXb);
    size_t qkvBytes = 3 * szXb;
    size_t hBytes   = (size_t)BL * DFFN * 2;
    size_t o_qkv  = alloc(qkvBytes > hBytes ? qkvBytes : hBytes);
    size_t o_ob   = alloc(szXb);
    size_t o_g32  = alloc(szXf);
    size_t o_wq   = alloc((size_t)3 * DMODEL * DMODEL * 2);
    size_t o_wk   = alloc((size_t)3 * DMODEL * DMODEL * 2);
    size_t o_wv   = alloc((size_t)3 * DMODEL * DMODEL * 2);
    size_t o_wo   = alloc((size_t)3 * DMODEL * DMODEL * 2);
    size_t o_w1   = alloc((size_t)3 * DMODEL * DFFN * 2);
    size_t o_w2   = alloc((size_t)3 * DMODEL * DFFN * 2);
    size_t o_wih  = alloc((size_t)4 * DRNN * DMODEL * 2);
    size_t o_wpr  = alloc((size_t)DMODEL * DRNN * 2);
    size_t o_wg   = alloc((size_t)DMODEL * DRNN * 2);
    size_t o_bsum = alloc(4 * DRNN * 4);
    size_t o_x0b  = alloc((size_t)BATCH * DMODEL * 2);
    size_t o_gts  = alloc((size_t)BATCH * 4 * DRNN * 4);
    size_t o_hhb  = alloc((size_t)BATCH * DRNN * 2);
    size_t o_outp = alloc((size_t)BATCH * DMODEL * 4);
    size_t o_outb = alloc((size_t)BATCH * DMODEL * 2);
    size_t o_enc  = alloc((size_t)BATCH * DRNN * 4);

    float* t_buf = (float*)(ws + o_t);
    float* x_f   = (float*)(ws + o_x);
    u16*   x_b   = (u16*)(ws + o_xb);
    u16*   q_b   = (u16*)(ws + o_qkv);
    u16*   k_b   = q_b + (size_t)BL * DMODEL;
    u16*   v_b   = k_b + (size_t)BL * DMODEL;
    u16*   h_b   = (u16*)(ws + o_qkv);
    u16*   o_b   = (u16*)(ws + o_ob);
    float* g32   = (float*)(ws + o_g32);

    auto convw = [&](const float* in, u16* outp, int K, int N, int tr) {
        int tot = K * N;
        k_conv_w<<<(tot + 255) / 256, 256, 0, stream>>>(in, outp, K, N, tr);
    };

    // weight conversion (transposed to N x K bf16)
    for (int l = 0; l < 3; ++l) {
        convw(Wq + (size_t)l * DMODEL * DMODEL, (u16*)(ws + o_wq) + (size_t)l * DMODEL * DMODEL, DMODEL, DMODEL, 1);
        convw(Wk + (size_t)l * DMODEL * DMODEL, (u16*)(ws + o_wk) + (size_t)l * DMODEL * DMODEL, DMODEL, DMODEL, 1);
        convw(Wv + (size_t)l * DMODEL * DMODEL, (u16*)(ws + o_wv) + (size_t)l * DMODEL * DMODEL, DMODEL, DMODEL, 1);
        convw(Wo + (size_t)l * DMODEL * DMODEL, (u16*)(ws + o_wo) + (size_t)l * DMODEL * DMODEL, DMODEL, DMODEL, 1);
        convw(W1 + (size_t)l * DMODEL * DFFN, (u16*)(ws + o_w1) + (size_t)l * DMODEL * DFFN, DMODEL, DFFN, 1);
        convw(W2 + (size_t)l * DMODEL * DFFN, (u16*)(ws + o_w2) + (size_t)l * DMODEL * DFFN, DFFN, DMODEL, 1);
    }
    convw(Wih,   (u16*)(ws + o_wih), DMODEL, 4 * DRNN, 0);   // already N x K
    convw(Wproj, (u16*)(ws + o_wpr), DRNN, DMODEL, 1);
    convw(Wg,    (u16*)(ws + o_wg),  DMODEL, DRNN, 1);

    k_cumsum_t<<<1, 32, 0, stream>>>(event_time, npm, t_buf);

    const float* xin = event_type;
    const int nPrep = (BL * DMODEL + 255) / 256;
    dim3 gSq(BL / 128, DMODEL / 128);
    dim3 gFf1(BL / 128, DFFN / 128);

    for (int l = 0; l < 3; ++l) {
        u16* WqT = (u16*)(ws + o_wq) + (size_t)l * DMODEL * DMODEL;
        u16* WkT = (u16*)(ws + o_wk) + (size_t)l * DMODEL * DMODEL;
        u16* WvT = (u16*)(ws + o_wv) + (size_t)l * DMODEL * DMODEL;
        u16* WoT = (u16*)(ws + o_wo) + (size_t)l * DMODEL * DMODEL;
        u16* W1T = (u16*)(ws + o_w1) + (size_t)l * DMODEL * DFFN;
        u16* W2T = (u16*)(ws + o_w2) + (size_t)l * DMODEL * DFFN;

        k_prep_x<<<nPrep, 256, 0, stream>>>(xin, t_buf, posv, npm, x_f, x_b);

        k_gemm_nt_big<<<gSq, 256, 0, stream>>>(x_b, DMODEL, WqT, DMODEL, nullptr, q_b,
                                               DMODEL, DMODEL, nullptr, 0);
        k_gemm_nt_big<<<gSq, 256, 0, stream>>>(x_b, DMODEL, WkT, DMODEL, nullptr, k_b,
                                               DMODEL, DMODEL, nullptr, 0);
        k_gemm_nt_big<<<gSq, 256, 0, stream>>>(x_b, DMODEL, WvT, DMODEL, nullptr, v_b,
                                               DMODEL, DMODEL, nullptr, 0);

        k_attn<<<dim3(SEQ / 16, NHEAD, BATCH), 128, 0, stream>>>(q_b, k_b, v_b, t_buf, o_b);

        k_gemm_nt_big<<<gSq, 256, 0, stream>>>(o_b, DMODEL, WoT, DMODEL, g32, nullptr,
                                               DMODEL, DMODEL, nullptr, 0);
        k_resid_ln<<<BL, 256, 0, stream>>>(x_f, g32, bo + l * DMODEL,
                                           ln1g + l * DMODEL, ln1b + l * DMODEL,
                                           npm, x_f, x_b);

        k_gemm_nt_big<<<gFf1, 256, 0, stream>>>(x_b, DMODEL, W1T, DMODEL, nullptr, h_b,
                                                DFFN, DMODEL, b1 + l * DFFN, 1);
        k_gemm_nt_big<<<gSq, 256, 0, stream>>>(h_b, DFFN, W2T, DFFN, g32, nullptr,
                                               DMODEL, DFFN, nullptr, 0);
        k_resid_ln<<<BL, 256, 0, stream>>>(x_f, g32, b2 + l * DMODEL,
                                           ln2g + l * DMODEL, ln2b + l * DMODEL,
                                           npm, x_f, x_b);
        xin = x_f;
    }

    // head
    k_x0_extract<<<(BATCH * DMODEL + 255) / 256, 256, 0, stream>>>(x_f, (u16*)(ws + o_x0b));
    k_bias_sum<<<(4 * DRNN + 255) / 256, 256, 0, stream>>>(bih, bhh, (float*)(ws + o_bsum), 4 * DRNN);

    k_gemm_nt<<<dim3(1, (4 * DRNN) / TN), 256, 0, stream>>>(
        (u16*)(ws + o_x0b), DMODEL, (u16*)(ws + o_wih), DMODEL,
        (float*)(ws + o_gts), nullptr, 4 * DRNN, BATCH, 4 * DRNN, DMODEL,
        (float*)(ws + o_bsum), 0);

    k_lstm<<<(BATCH * DRNN + 255) / 256, 256, 0, stream>>>((float*)(ws + o_gts), (u16*)(ws + o_hhb));

    k_gemm_nt<<<dim3(1, DMODEL / TN), 256, 0, stream>>>(
        (u16*)(ws + o_hhb), DRNN, (u16*)(ws + o_wpr), DRNN,
        (float*)(ws + o_outp), (u16*)(ws + o_outb), DMODEL, BATCH, DMODEL, DRNN,
        bproj, 0);

    k_gemm_nt<<<dim3(1, DRNN / TN), 256, 0, stream>>>(
        (u16*)(ws + o_outb), DMODEL, (u16*)(ws + o_wg), DMODEL,
        (float*)(ws + o_enc), nullptr, DRNN, BATCH, DRNN, DMODEL,
        bg, 0);

    k_unsort<<<(BATCH * DRNN + 255) / 256, 256, 0, stream>>>(
        (float*)(ws + o_enc), sort_idx, (float*)d_out);
}
